// TextRNN_24404004176160
// MI455X (gfx1250) — matrix operand order repaired and checked by reading, not compiled, not run on hardware
//
#include <hip/hip_runtime.h>
#include <hip/hip_bf16.h>

// ---------------------------------------------------------------------------
// TextRNN forward for MI455X (gfx1250, wave32, WMMA).
//   emb = emb_W[X]                         [S,B,D]
//   pre0 = emb @ W_ih0^T  (parallel GEMM, removes ih-part from recurrence)
//   2x tanh-RNN layers (H=200)             sequential, 1 WG, 13 waves, no spill
//   decode = y1 @ dec_W^T + dec_b          [S,B,V] dominant: 823MB NT stores,
//                                          16x48 register tile per wave
// Matrix math in bf16 WMMA (v_wmma_f32_16x16x32_bf16) with f32 accumulate.
// ---------------------------------------------------------------------------

#define V_  50257
#define D_  200
#define H_  200
#define S_  128
#define B_  32
#define KP  224          // K padded to 7 tiles of 32
#define HP  208          // H padded to 13 tiles of 16
#define VP  50272        // V padded to 3142 tiles of 16
#define NTILES 3142      // ceil(V/16)

typedef __attribute__((ext_vector_type(16))) __bf16 v16bf;
typedef __attribute__((ext_vector_type(8)))  __bf16 v8bf;
typedef __attribute__((ext_vector_type(8)))  float  v8f;

// A-fragment (16x32 bf16, row-major source, ld = KP):
// lane 0-15 : row = lane,    K = k0+{0..7}  and k0+{16..23}
// lane16-31 : row = lane-16, K = k0+{8..15} and k0+{24..31}
__device__ __forceinline__ v16bf load_a_frag(const __bf16* base, int row0,
                                             int k0, int lane) {
    int r   = row0 + (lane & 15);
    int klo = k0 + ((lane & 16) ? 8 : 0);
    const __bf16* p = base + (size_t)r * KP + klo;
    union { v16bf v; v8bf h[2]; } u;
    u.h[0] = *(const v8bf*)(p);
    u.h[1] = *(const v8bf*)(p + 16);
    return u.v;
}

// B-fragment (32x16 bf16). Source stored [n][k] row-major, ld = KP.
// lane 0-15 : col = lane,    K = k0+{0..7}  and k0+{16..23}
// lane16-31 : col = lane-16, K = k0+{8..15} and k0+{24..31}
__device__ __forceinline__ v16bf load_b_frag(const __bf16* base, int col0,
                                             int k0, int lane) {
    int n = col0 + (lane & 15);
    int k = k0 + ((lane & 16) ? 8 : 0);
    const __bf16* p = base + (size_t)n * KP + k;
    union { v16bf v; v8bf h[2]; } u;
    u.h[0] = *(const v8bf*)(p);
    u.h[1] = *(const v8bf*)(p + 16);
    return u.v;
}

__device__ __forceinline__ v8f wmma_bf16(v16bf a, v16bf b, v8f c) {
    return __builtin_amdgcn_wmma_f32_16x16x32_bf16(
        /*neg_a=*/false, a, /*neg_b=*/false, b,
        /*c_mod=*/(short)0, c, /*reuse_a=*/false, /*reuse_b=*/false);
}

// ---- prep: RNN weights f32[200x200] -> bf16[208x224] zero-padded ----------
__global__ void k_wconv(const float* __restrict__ w0, const float* __restrict__ w1,
                        const float* __restrict__ w2, const float* __restrict__ w3,
                        __bf16* __restrict__ dst) {
    int row = blockIdx.x;          // 0..207
    int w   = blockIdx.y;          // 0..3
    int col = threadIdx.x;         // 0..223
    const float* src = (w == 0) ? w0 : (w == 1) ? w1 : (w == 2) ? w2 : w3;
    float v = (row < H_ && col < D_) ? src[row * D_ + col] : 0.f;
    dst[((size_t)w * HP + row) * KP + col] = (__bf16)v;
}

// ---- prep: dec_W f32[V x 200] -> bf16[VP x 224] zero-padded ---------------
__global__ void k_decwconv(const float* __restrict__ decw, __bf16* __restrict__ dst) {
    int row = blockIdx.x;          // 0..VP-1
    int col = threadIdx.x;         // 0..223
    float v = (row < V_ && col < H_) ? decw[(size_t)row * H_ + col] : 0.f;
    dst[(size_t)row * KP + col] = (__bf16)v;
}

// ---- prep: embedding gather to bf16 [S*B x 224]; zero y1 pad cols ---------
__global__ void k_embed(const int* __restrict__ X, const float* __restrict__ embW,
                        __bf16* __restrict__ emb, __bf16* __restrict__ y1) {
    int sb  = blockIdx.x;          // 0..4095  (= s*32 + b)
    int t   = threadIdx.x;         // 0..223
    int tok = X[sb];
    float v = (t < D_) ? embW[(size_t)tok * D_ + t] : 0.f;
    emb[(size_t)sb * KP + t] = (__bf16)v;
    if (t >= HP) y1[(size_t)sb * KP + t] = (__bf16)0.f;  // cols 208..223
}

// ---- parallel: pre0 = emb @ Wih0^T  -> f32 [4096 x 208] -------------------
// One wave per (m-tile, n-tile); 3328 waves; removes the input-side GEMM from
// the sequential recurrence.
__global__ __launch_bounds__(128) void
k_pre0(const __bf16* __restrict__ emb, const __bf16* __restrict__ Wih0,
       float* __restrict__ pre0) {
    int wave = threadIdx.x >> 5, lane = threadIdx.x & 31;
    int wid  = blockIdx.x * 4 + wave;        // 0..3327
    int m = wid / 13, nt = wid % 13;
    int n0 = nt * 16;

    v8f c = {};
    const __bf16* arow = emb + (size_t)m * 16 * KP;
#pragma unroll
    for (int kt = 0; kt < 7; ++kt) {
        v16bf a = load_a_frag(arow, 0, kt * 32, lane);
        v16bf b = load_b_frag(Wih0, n0, kt * 32, lane);
        c = wmma_bf16(a, b, c);
    }
    int col   = n0 + (lane & 15);
    int rbase = m * 16 + ((lane & 16) ? 8 : 0);
#pragma unroll
    for (int r = 0; r < 8; ++r)
        pre0[(size_t)(rbase + r) * HP + col] = c[r];
}

// ---- persistent 2-layer tanh RNN: one WG, 13 waves, each wave = 1 N-tile,
//      both M-tiles; Whh0/Wih1/Whh1 fragments register-resident (168 VGPRs),
//      layer-0 accumulators seeded from pre0 ------------------------------
__global__ __launch_bounds__(416) void
k_rnn(const float* __restrict__ pre0, const __bf16* __restrict__ Wall,
      const float* __restrict__ hidden_in, __bf16* __restrict__ y1,
      float* __restrict__ hid_out) {
    __shared__ __bf16 h0[B_ * KP];
    __shared__ __bf16 h1[B_ * KP];
    __shared__ __bf16 y0[B_ * KP];

    const __bf16* Whh0 = Wall + (size_t)1 * HP * KP;
    const __bf16* Wih1 = Wall + (size_t)2 * HP * KP;
    const __bf16* Whh1 = Wall + (size_t)3 * HP * KP;

    int tid = threadIdx.x;
    for (int i = tid; i < B_ * KP; i += 416) {
        int b = i / KP, c = i % KP;
        h0[i] = (c < H_) ? (__bf16)hidden_in[0 * B_ * H_ + b * H_ + c] : (__bf16)0.f;
        h1[i] = (c < H_) ? (__bf16)hidden_in[1 * B_ * H_ + b * H_ + c] : (__bf16)0.f;
        y0[i] = (__bf16)0.f;
    }
    __syncthreads();

    int nt = tid >> 5, lane = tid & 31;     // 13 waves = 13 N-tiles
    int n0 = nt * 16;
    int col  = n0 + (lane & 15);
    int roff = (lane & 16) ? 8 : 0;

    // preload weight fragments (3 matrices x 7 k-tiles = 168 VGPRs)
    v16bf bhh0[7], bih1[7], bhh1[7];
#pragma unroll
    for (int kt = 0; kt < 7; ++kt) {
        bhh0[kt] = load_b_frag(Whh0, n0, kt * 32, lane);
        bih1[kt] = load_b_frag(Wih1, n0, kt * 32, lane);
        bhh1[kt] = load_b_frag(Whh1, n0, kt * 32, lane);
    }

    for (int s = 0; s < S_; ++s) {
        // ----- layer 0: h0_new = tanh(pre0[s] + h0 @ Whh0^T) -----
        v8f cA, cB;
        const float* prow = pre0 + (size_t)s * B_ * HP;
#pragma unroll
        for (int r = 0; r < 8; ++r) {
            cA[r] = prow[(size_t)(roff + r) * HP + col];
            cB[r] = prow[(size_t)(16 + roff + r) * HP + col];
        }
#pragma unroll
        for (int kt = 0; kt < 7; ++kt) {
            v16bf a0 = load_a_frag(h0,  0, kt * 32, lane);
            v16bf a1 = load_a_frag(h0, 16, kt * 32, lane);
            cA = wmma_bf16(a0, bhh0[kt], cA);
            cB = wmma_bf16(a1, bhh0[kt], cB);
        }
        __syncthreads();   // all reads of h0 done before overwrite
#pragma unroll
        for (int r = 0; r < 8; ++r) {
            float tA = tanhf(cA[r]);
            float tB = tanhf(cB[r]);
            int rowA = roff + r, rowB = 16 + roff + r;
            __bf16 a = (__bf16)tA, b = (__bf16)tB;
            h0[rowA * KP + col] = a;  y0[rowA * KP + col] = a;
            h0[rowB * KP + col] = b;  y0[rowB * KP + col] = b;
            if (s == S_ - 1 && col < H_) {
                hid_out[0 * B_ * H_ + rowA * H_ + col] = tA;
                hid_out[0 * B_ * H_ + rowB * H_ + col] = tB;
            }
        }
        __syncthreads();

        // ----- layer 1: h1_new = tanh(y0 @ Wih1^T + h1 @ Whh1^T) -----
        v8f dA = {}, dB = {};
#pragma unroll
        for (int kt = 0; kt < 7; ++kt) {
            v16bf a0 = load_a_frag(y0,  0, kt * 32, lane);
            v16bf a1 = load_a_frag(y0, 16, kt * 32, lane);
            dA = wmma_bf16(a0, bih1[kt], dA);
            dB = wmma_bf16(a1, bih1[kt], dB);
        }
#pragma unroll
        for (int kt = 0; kt < 7; ++kt) {
            v16bf a0 = load_a_frag(h1,  0, kt * 32, lane);
            v16bf a1 = load_a_frag(h1, 16, kt * 32, lane);
            dA = wmma_bf16(a0, bhh1[kt], dA);
            dB = wmma_bf16(a1, bhh1[kt], dB);
        }
        __syncthreads();
#pragma unroll
        for (int r = 0; r < 8; ++r) {
            float tA = tanhf(dA[r]);
            float tB = tanhf(dB[r]);
            int rowA = roff + r, rowB = 16 + roff + r;
            __bf16 a = (__bf16)tA, b = (__bf16)tB;
            h1[rowA * KP + col] = a;
            h1[rowB * KP + col] = b;
            y1[((size_t)s * B_ + rowA) * KP + col] = a;
            y1[((size_t)s * B_ + rowB) * KP + col] = b;
            if (s == S_ - 1 && col < H_) {
                hid_out[1 * B_ * H_ + rowA * H_ + col] = tA;
                hid_out[1 * B_ * H_ + rowB * H_ + col] = tB;
            }
        }
        __syncthreads();
    }
}

// ---- decode: [4096 x 224] x [VP x 224]^T -> [4096 x V] f32 ----------------
// One wave computes a 16x48 strip (3 N-tiles, 21 B-frags register-resident),
// loops all 256 M-tiles; each A fragment feeds 3 WMMAs (3x less L2 traffic).
// Non-temporal stores for the 823MB output stream.
__global__ __launch_bounds__(128) void
k_decode(const __bf16* __restrict__ y1, const __bf16* __restrict__ decw,
         const float* __restrict__ dec_b, float* __restrict__ out) {
    int wave = threadIdx.x >> 5, lane = threadIdx.x & 31;
    int wid  = blockIdx.x * 4 + wave;
    int base = wid * 3;
    if (base >= NTILES) return;
    int nvalid = NTILES - base; if (nvalid > 3) nvalid = 3;   // wave-uniform

    v16bf bf[3][7];
    float bias[3];
    int   vcol[3];
#pragma unroll
    for (int t = 0; t < 3; ++t) {
        int tile = (base + t < NTILES) ? (base + t) : base;   // clamp: no OOB
        int v0 = tile * 16;
#pragma unroll
        for (int kt = 0; kt < 7; ++kt)
            bf[t][kt] = load_b_frag(decw, v0, kt * 32, lane);
        vcol[t] = v0 + (lane & 15);
        bias[t] = (vcol[t] < V_) ? dec_b[vcol[t]] : 0.f;
    }
    int roff = (lane & 16) ? 8 : 0;

    for (int m = 0; m < (S_ * B_) / 16; ++m) {
        v8f c0 = {}, c1 = {}, c2 = {};
        const __bf16* arow = y1 + (size_t)m * 16 * KP;
#pragma unroll
        for (int kt = 0; kt < 7; ++kt) {
            v16bf a = load_a_frag(arow, 0, kt * 32, lane);
            c0 = wmma_bf16(a, bf[0][kt], c0);
            c1 = wmma_bf16(a, bf[1][kt], c1);
            c2 = wmma_bf16(a, bf[2][kt], c2);
        }
        int rbase = m * 16 + roff;
        if (vcol[0] < V_) {
#pragma unroll
            for (int r = 0; r < 8; ++r)
                __builtin_nontemporal_store(c0[r] + bias[0],
                    &out[(size_t)(rbase + r) * V_ + vcol[0]]);
        }
        if (nvalid > 1 && vcol[1] < V_) {
#pragma unroll
            for (int r = 0; r < 8; ++r)
                __builtin_nontemporal_store(c1[r] + bias[1],
                    &out[(size_t)(rbase + r) * V_ + vcol[1]]);
        }
        if (nvalid > 2 && vcol[2] < V_) {
#pragma unroll
            for (int r = 0; r < 8; ++r)
                __builtin_nontemporal_store(c2[r] + bias[2],
                    &out[(size_t)(rbase + r) * V_ + vcol[2]]);
        }
    }
}

// ---------------------------------------------------------------------------
extern "C" void kernel_launch(void* const* d_in, const int* in_sizes, int n_in,
                              void* d_out, int out_size, void* d_ws, size_t ws_size,
                              hipStream_t stream) {
    const int*   X      = (const int*)  d_in[0];
    const float* hidden = (const float*)d_in[1];
    const float* embW   = (const float*)d_in[2];
    const float* Wih0   = (const float*)d_in[3];
    const float* Whh0   = (const float*)d_in[4];
    const float* Wih1   = (const float*)d_in[5];
    const float* Whh1   = (const float*)d_in[6];
    const float* decW   = (const float*)d_in[7];
    const float* decb   = (const float*)d_in[8];
    float* out = (float*)d_out;

    // workspace layout (bytes, all 256-aligned)
    char* ws = (char*)d_ws;
    __bf16* emb_bf  = (__bf16*)(ws + 0);          // 4096*224*2 = 1,835,008
    __bf16* W_bf    = (__bf16*)(ws + 1835008);    // 4*208*224*2 = 372,736
    __bf16* y1_bf   = (__bf16*)(ws + 2207744);    // 4096*224*2 = 1,835,008
    __bf16* decw_bf = (__bf16*)(ws + 4042752);    // 50272*224*2 = 22,521,856
    float*  pre0    = (float*) (ws + 26564608);   // 4096*208*4 = 3,407,872
    // total ~30 MB

    float* hid_out = out + (size_t)S_ * B_ * V_;  // hidden block after decode

    int nwaves  = (NTILES + 2) / 3;               // 1048 decode waves
    int nblocks = (nwaves + 3) / 4;               // 4 waves per block

    k_wconv   <<<dim3(HP, 4), KP, 0, stream>>>(Wih0, Whh0, Wih1, Whh1, W_bf);
    k_decwconv<<<VP,          KP, 0, stream>>>(decW, decw_bf);
    k_embed   <<<S_ * B_,     KP, 0, stream>>>(X, embW, emb_bf, y1_bf);
    k_pre0    <<<832,         128, 0, stream>>>(emb_bf, W_bf, pre0);
    k_rnn     <<<1,           416, 0, stream>>>(pre0, W_bf, hidden, y1_bf, hid_out);
    k_decode  <<<nblocks,     128, 0, stream>>>(y1_bf, decw_bf, decb, out);
}
